// PreludeBlock_76261439308445
// MI455X (gfx1250) — matrix operand order repaired and checked by reading, not compiled, not run on hardware
//
#include <hip/hip_runtime.h>
#include <cstdint>

// ---------------------------------------------------------------------------
// Mamba-like block on MI455X (gfx1250): all large GEMMs via
// v_wmma_f32_16x16x32_bf16 (fp32 accumulate), async global->LDS staging
// (global_load_async_to_lds_b128 / ASYNCcnt) with double buffering;
// scan/norms in fp32 VALU.
// ---------------------------------------------------------------------------

typedef __bf16 bf16;
typedef __attribute__((ext_vector_type(16))) __bf16 v16bf;
typedef __attribute__((ext_vector_type(8)))  __bf16 v8bf;
typedef __attribute__((ext_vector_type(8)))  float  v8f;

#define DMODEL 1024
#define DINNER 2048
#define DSTATE 16
#define FFNH   2752
#define SEQL   1024
#define BATCH  2
#define MTOK   (BATCH * SEQL)   // 2048 tokens

// ---- workspace layout (bytes, all 256-aligned) ----------------------------
static constexpr size_t O_WIN  = 0;                       // in_proj  bf16  2048x1024
static constexpr size_t O_WGT  = O_WIN  + 4194304;        // gate     bf16  2048x1024
static constexpr size_t O_WDT  = O_WGT  + 4194304;        // dt_w     bf16  2048x2048
static constexpr size_t O_WOUT = O_WDT  + 8388608;        // out_proj bf16  1024x2048
static constexpr size_t O_WFG  = O_WOUT + 4194304;        // ffn_gate bf16  2752x1024
static constexpr size_t O_WFU  = O_WFG  + 5636096;        // ffn_up   bf16  2752x1024
static constexpr size_t O_WFD  = O_WFU  + 5636096;        // ffn_down bf16  1024x2752
static constexpr size_t O_HB   = O_WFD  + 5636096;        // h bf16 2048x1024 (reused as hn)
static constexpr size_t O_ZRAW = O_HB   + 4194304;        // z_raw f32 2048x2048 (reused as yb bf16)
static constexpr size_t O_GRAW = O_ZRAW + 16777216;       // gate_raw f32 (reused as dt f32)
static constexpr size_t O_ZF   = O_GRAW + 16777216;       // gated z f32 2048x2048
static constexpr size_t O_ZB   = O_ZF   + 16777216;       // gated z bf16
static constexpr size_t O_BC   = O_ZB   + 8388608;        // B,C f32 2048x32
static constexpr size_t O_X2   = O_BC   + 262144;         // x + ssm residual f32 2048x1024
static constexpr size_t O_GF   = O_X2   + 8388608;        // ffn gate f32 2048x2752
static constexpr size_t O_UF   = O_GF   + 22544384;       // ffn up   f32 2048x2752
static constexpr size_t O_TB   = O_UF   + 22544384;       // silu(g)*u bf16 2048x2752

// ---- CDNA5 async global->LDS path (inline asm, ASYNCcnt-tracked) ----------
#define USE_ASYNC_LDS 1

__device__ __forceinline__ void async_cp16(const bf16* g, bf16* l) {
#ifdef USE_ASYNC_LDS
  unsigned lds = (unsigned)(uintptr_t)l;           // LDS byte offset (addr[31:0])
  unsigned long long ga = (unsigned long long)(uintptr_t)g;
  asm volatile("global_load_async_to_lds_b128 %0, %1, off"
               :: "v"(lds), "v"(ga) : "memory");
#else
  *reinterpret_cast<uint4*>(l) = *reinterpret_cast<const uint4*>(g);
#endif
}

template<int N>
__device__ __forceinline__ void async_wait() {
#ifdef USE_ASYNC_LDS
  asm volatile("s_wait_asynccnt %0" :: "i"(N) : "memory");
#endif
}

// ---------------------------------------------------------------------------
__device__ __forceinline__ bf16 f2bf(float f) {
  union { float f; unsigned u; } a; a.f = f;
  unsigned r = a.u + 0x7FFFu + ((a.u >> 16) & 1u);   // round-to-nearest-even
  unsigned short h = (unsigned short)(r >> 16);
  return __builtin_bit_cast(bf16, h);
}

__global__ __launch_bounds__(256)
void cast_f32_bf16(const float* __restrict__ in, bf16* __restrict__ out, int n) {
  int i = blockIdx.x * 256 + threadIdx.x;
  if (i < n) out[i] = f2bf(in[i]);
}

__global__ __launch_bounds__(256)
void rmsnorm_bf16(const float* __restrict__ x, const float* __restrict__ w,
                  bf16* __restrict__ out, int D) {
  __shared__ float red[8];
  int row = blockIdx.x;
  const float* xr = x + (size_t)row * D;
  float ss = 0.f;
  for (int i = threadIdx.x; i < D; i += 256) { float v = xr[i]; ss += v * v; }
  for (int off = 16; off > 0; off >>= 1) ss += __shfl_down(ss, off);
  int lane = threadIdx.x & 31, wave = threadIdx.x >> 5;
  if (lane == 0) red[wave] = ss;
  __syncthreads();
  if (threadIdx.x == 0) {
    float t = 0.f;
    for (int i = 0; i < 8; ++i) t += red[i];
    red[0] = rsqrtf(t / (float)D + 1e-6f);
  }
  __syncthreads();
  float s = red[0];
  for (int i = threadIdx.x; i < D; i += 256)
    out[(size_t)row * D + i] = f2bf(xr[i] * s * w[i]);
}

// ---------------------------------------------------------------------------
// WMMA GEMM: C[M,N] = A[M,K](bf16,row) * W[N,K](bf16,row)^T (+ Res)
// 256 threads = 8 wave32 in a 4x2 grid; block tile 256x64, wave tile 64x32,
// BK=32, double-buffered LDS filled with global_load_async_to_lds_b128.
// ---------------------------------------------------------------------------
#define BM 256
#define BN 64
#define BK 32
#define LSTR 40   // bf16 per LDS row (32 + 8 pad) -> 80B rows, 16B aligned

template<bool ADD_RES>
__global__ __launch_bounds__(256)
void gemm_bf16_wmma(const bf16* __restrict__ A, const bf16* __restrict__ W,
                    const float* __restrict__ Res, float* __restrict__ C,
                    int M, int N, int K) {
  __shared__ alignas(16) bf16 As[2][BM * LSTR];
  __shared__ alignas(16) bf16 Ws[2][BN * LSTR];
  const int tid  = threadIdx.x;
  const int wave = tid >> 5, lane = tid & 31;
  const int hi = lane >> 4, lr = lane & 15;
  const int m0 = blockIdx.y * BM, n0 = blockIdx.x * BN;
  const int wm = (wave & 3) * 64;    // 4 m-waves x 64 rows
  const int wn = (wave >> 2) * 32;   // 2 n-waves x 32 cols

  // ---- per-thread copy assignments (uniform, no divergence) ----
  // A: 1024 16B-chunks/tile -> 4 per thread (rows r, r+64, r+128, r+192)
  // W:  256 16B-chunks/tile -> 1 per thread
  const int r = tid >> 2, q = tid & 3;
  const bf16* gA = A + (size_t)(m0 + r) * K + q * 8;
  const bf16* gW = W + (size_t)(n0 + r) * K + q * 8;
  const size_t a64 = (size_t)64 * K;
  const int ldsA = r * LSTR + q * 8;          // element offset in buffer
  const int ldsW = r * LSTR + q * 8;

  v8f acc[4][2];
#pragma unroll
  for (int i = 0; i < 4; ++i)
#pragma unroll
    for (int j = 0; j < 2; ++j)
#pragma unroll
      for (int v = 0; v < 8; ++v) acc[i][j][v] = 0.f;

  auto issue = [&](int buf) {
    bf16* dA = &As[buf][ldsA];
    async_cp16(gA,           dA);
    async_cp16(gA + a64,     dA + 64 * LSTR);
    async_cp16(gA + 2 * a64, dA + 128 * LSTR);
    async_cp16(gA + 3 * a64, dA + 192 * LSTR);
    async_cp16(gW, &Ws[buf][ldsW]);
    gA += BK;
    gW += BK;
  };

  issue(0);
  const int KT = K / BK;
  for (int t = 0; t < KT; ++t) {
    const int buf = t & 1;
    if (t + 1 < KT) {
      issue(buf ^ 1);
      async_wait<5>();     // tile t's 5 per-wave async ops complete
    } else {
      async_wait<0>();
    }
    __syncthreads();

    const bf16* cA = As[buf];
    const bf16* cW = Ws[buf];

    // A fragment (16x32): m = lr; e<8 -> k = hi*8+e ; e>=8 -> k = 16+hi*8+(e-8)
    v16bf af[4];
#pragma unroll
    for (int i = 0; i < 4; ++i) {
      int row = wm + i * 16 + lr;
      v8bf lo  = *reinterpret_cast<const v8bf*>(&cA[row * LSTR + hi * 8]);
      v8bf hi8 = *reinterpret_cast<const v8bf*>(&cA[row * LSTR + 16 + hi * 8]);
      af[i] = __builtin_shufflevector(lo, hi8, 0,1,2,3,4,5,6,7,8,9,10,11,12,13,14,15);
    }
    // B fragment (32x16): n = lr; e<8 -> k = hi*8+e ; e>=8 -> k = 16+hi*8+(e-8)   (B[k][n] = W[n][k])
    v16bf bfr[2];
#pragma unroll
    for (int j = 0; j < 2; ++j) {
      int row = wn + j * 16 + lr;
      v8bf lo  = *reinterpret_cast<const v8bf*>(&cW[row * LSTR + hi * 8]);
      v8bf hi8 = *reinterpret_cast<const v8bf*>(&cW[row * LSTR + 16 + hi * 8]);
      bfr[j] = __builtin_shufflevector(lo, hi8, 0,1,2,3,4,5,6,7,8,9,10,11,12,13,14,15);
    }

#pragma unroll
    for (int i = 0; i < 4; ++i)
#pragma unroll
      for (int j = 0; j < 2; ++j)
        acc[i][j] = __builtin_amdgcn_wmma_f32_16x16x32_bf16(
            false, af[i], false, bfr[j], (short)0, acc[i][j], false, false);

    __syncthreads();   // all waves done reading buf before it is refilled
  }

  // D layout: m = base + hi*8 + v, n = base + lr
#pragma unroll
  for (int i = 0; i < 4; ++i)
#pragma unroll
    for (int j = 0; j < 2; ++j) {
      int mbase = m0 + wm + i * 16 + hi * 8;
      int n = n0 + wn + j * 16 + lr;
#pragma unroll
      for (int v = 0; v < 8; ++v) {
        size_t idx = (size_t)(mbase + v) * N + n;
        float val = acc[i][j][v];
        if (ADD_RES) val += Res[idx];
        C[idx] = val;
      }
    }
}

// ---------------------------------------------------------------------------
__global__ __launch_bounds__(256)
void gate_mul(const float* __restrict__ zraw, const float* __restrict__ graw,
              float* __restrict__ zf, bf16* __restrict__ zb, int n) {
  int i = blockIdx.x * 256 + threadIdx.x;
  if (i < n) {
    float g = 1.f / (1.f + __expf(-graw[i]));
    float z = zraw[i] * g;
    zf[i] = z;
    zb[i] = f2bf(z);
  }
}

__global__ __launch_bounds__(256)
void softplus_bias(float* __restrict__ dt, const float* __restrict__ b,
                   int n, int colmask) {
  int i = blockIdx.x * 256 + threadIdx.x;
  if (i < n) {
    float v = dt[i] + b[i & colmask];
    dt[i] = (v > 20.f) ? v : log1pf(__expf(v));
  }
}

__global__ __launch_bounds__(256)
void silu_mul(const float* __restrict__ g, const float* __restrict__ u,
              bf16* __restrict__ t, int n) {
  int i = blockIdx.x * 256 + threadIdx.x;
  if (i < n) {
    float gv = g[i];
    float s = gv / (1.f + __expf(-gv));
    t[i] = f2bf(s * u[i]);
  }
}

// BC[m, 0..31] = sum_k z[m,k] * x_proj_w[j,k]; each wave owns 4 outputs,
// lanes stride k (coalesced), shuffle-reduce.
__global__ __launch_bounds__(256)
void xproj_kernel(const float* __restrict__ z, const float* __restrict__ w,
                  float* __restrict__ bc, int K) {
  int m = blockIdx.x;
  int wave = threadIdx.x >> 5, lane = threadIdx.x & 31;
  const float* zr = z + (size_t)m * K;
  const float* w0 = w + (size_t)(wave * 4 + 0) * K;
  const float* w1 = w + (size_t)(wave * 4 + 1) * K;
  const float* w2 = w + (size_t)(wave * 4 + 2) * K;
  const float* w3 = w + (size_t)(wave * 4 + 3) * K;
  float a0 = 0.f, a1 = 0.f, a2 = 0.f, a3 = 0.f;
  for (int k = lane; k < K; k += 32) {
    float zv = zr[k];
    a0 += zv * w0[k]; a1 += zv * w1[k]; a2 += zv * w2[k]; a3 += zv * w3[k];
  }
  for (int off = 16; off > 0; off >>= 1) {
    a0 += __shfl_down(a0, off); a1 += __shfl_down(a1, off);
    a2 += __shfl_down(a2, off); a3 += __shfl_down(a3, off);
  }
  if (lane == 0) {
    float* o = bc + (size_t)m * 32 + wave * 4;
    o[0] = a0; o[1] = a1; o[2] = a2; o[3] = a3;
  }
}

// Selective scan: one thread per (b, d) chain, state in registers,
// B/C staged in LDS in 64-timestep chunks; dt/z loads lane-coalesced over d.
__global__ __launch_bounds__(256)
void scan_kernel(const float* __restrict__ dt, const float* __restrict__ z,
                 const float* __restrict__ bcg, const float* __restrict__ A_log,
                 const float* __restrict__ Dp, bf16* __restrict__ yb) {
  int b = blockIdx.y;
  int d = blockIdx.x * 256 + threadIdx.x;
  float a[DSTATE], h[DSTATE];
#pragma unroll
  for (int n = 0; n < DSTATE; ++n) {
    a[n] = -__expf(A_log[(size_t)d * DSTATE + n]);
    h[n] = 0.f;
  }
  float dp = Dp[d];
  __shared__ float bc[64 * 32];
  for (int l0 = 0; l0 < SEQL; l0 += 64) {
    __syncthreads();
    for (int i = threadIdx.x; i < 64 * 32; i += 256)
      bc[i] = bcg[((size_t)b * SEQL + l0) * 32 + i];
    __syncthreads();
    for (int li = 0; li < 64; ++li) {
      size_t idx = ((size_t)b * SEQL + l0 + li) * DINNER + d;
      float dtv = dt[idx], zv = z[idx];
      float dz = dtv * zv;
      float y = zv * dp;
#pragma unroll
      for (int n = 0; n < DSTATE; ++n) {
        float da = __expf(dtv * a[n]);
        h[n] = da * h[n] + dz * bc[li * 32 + n];
        y += h[n] * bc[li * 32 + 16 + n];
      }
      yb[idx] = f2bf(y);
    }
  }
}

// ---------------------------------------------------------------------------
extern "C" void kernel_launch(void* const* d_in, const int* in_sizes, int n_in,
                              void* d_out, int out_size, void* d_ws, size_t ws_size,
                              hipStream_t stream) {
  (void)in_sizes; (void)n_in; (void)out_size; (void)ws_size;
  const float* x         = (const float*)d_in[0];
  const float* norm1_w   = (const float*)d_in[1];
  const float* in_proj   = (const float*)d_in[2];
  const float* gate_proj = (const float*)d_in[3];
  const float* dt_w      = (const float*)d_in[4];
  const float* dt_b      = (const float*)d_in[5];
  const float* x_proj_w  = (const float*)d_in[6];
  const float* A_log     = (const float*)d_in[7];
  const float* Dp        = (const float*)d_in[8];
  const float* out_proj  = (const float*)d_in[9];
  const float* ffn_norm  = (const float*)d_in[10];
  const float* ffn_gate  = (const float*)d_in[11];
  const float* ffn_up    = (const float*)d_in[12];
  const float* ffn_down  = (const float*)d_in[13];
  float* out = (float*)d_out;
  char* ws = (char*)d_ws;

  bf16* w_in  = (bf16*)(ws + O_WIN);
  bf16* w_gt  = (bf16*)(ws + O_WGT);
  bf16* w_dt  = (bf16*)(ws + O_WDT);
  bf16* w_out = (bf16*)(ws + O_WOUT);
  bf16* w_fg  = (bf16*)(ws + O_WFG);
  bf16* w_fu  = (bf16*)(ws + O_WFU);
  bf16* w_fd  = (bf16*)(ws + O_WFD);
  bf16*  hb    = (bf16*)(ws + O_HB);      // also hn
  float* zraw  = (float*)(ws + O_ZRAW);
  bf16*  yb    = (bf16*)(ws + O_ZRAW);    // reuse after zraw dead
  float* graw  = (float*)(ws + O_GRAW);
  float* dtbuf = (float*)(ws + O_GRAW);   // reuse after gate consumed
  float* zf    = (float*)(ws + O_ZF);
  bf16*  zb    = (bf16*)(ws + O_ZB);
  float* bcbuf = (float*)(ws + O_BC);
  float* x2    = (float*)(ws + O_X2);
  float* gf    = (float*)(ws + O_GF);
  float* uf    = (float*)(ws + O_UF);
  bf16*  tb    = (bf16*)(ws + O_TB);

  auto cast = [&](const float* src, bf16* dst, int n) {
    cast_f32_bf16<<<(n + 255) / 256, 256, 0, stream>>>(src, dst, n);
  };
  cast(in_proj,   w_in,  DINNER * DMODEL);
  cast(gate_proj, w_gt,  DINNER * DMODEL);
  cast(dt_w,      w_dt,  DINNER * DINNER);
  cast(out_proj,  w_out, DMODEL * DINNER);
  cast(ffn_gate,  w_fg,  FFNH * DMODEL);
  cast(ffn_up,    w_fu,  FFNH * DMODEL);
  cast(ffn_down,  w_fd,  DMODEL * FFNH);

  dim3 blk(256);

  // 1) h = rmsnorm(x) -> bf16
  rmsnorm_bf16<<<MTOK, blk, 0, stream>>>(x, norm1_w, hb, DMODEL);

  // 2) z_raw = h @ in_proj^T ; g_raw = h @ gate_proj^T  (WMMA bf16)
  gemm_bf16_wmma<false><<<dim3(DINNER / BN, MTOK / BM), blk, 0, stream>>>(
      hb, w_in, nullptr, zraw, MTOK, DINNER, DMODEL);
  gemm_bf16_wmma<false><<<dim3(DINNER / BN, MTOK / BM), blk, 0, stream>>>(
      hb, w_gt, nullptr, graw, MTOK, DINNER, DMODEL);

  // 3) z = z_raw * sigmoid(g_raw)
  gate_mul<<<(MTOK * DINNER + 255) / 256, blk, 0, stream>>>(
      zraw, graw, zf, zb, MTOK * DINNER);

  // 4) dt = softplus(z @ dt_w^T + dt_b)
  gemm_bf16_wmma<false><<<dim3(DINNER / BN, MTOK / BM), blk, 0, stream>>>(
      zb, w_dt, nullptr, dtbuf, MTOK, DINNER, DINNER);
  softplus_bias<<<(MTOK * DINNER + 255) / 256, blk, 0, stream>>>(
      dtbuf, dt_b, MTOK * DINNER, DINNER - 1);

  // 5) BC = z @ x_proj^T ; selective scan -> y (bf16)
  xproj_kernel<<<MTOK, blk, 0, stream>>>(zf, x_proj_w, bcbuf, DINNER);
  scan_kernel<<<dim3(DINNER / 256, BATCH), blk, 0, stream>>>(
      dtbuf, zf, bcbuf, A_log, Dp, yb);

  // 6) x2 = x + y @ out_proj^T  (residual fused)
  gemm_bf16_wmma<true><<<dim3(DMODEL / BN, MTOK / BM), blk, 0, stream>>>(
      yb, w_out, x, x2, MTOK, DMODEL, DINNER);

  // 7) FFN
  rmsnorm_bf16<<<MTOK, blk, 0, stream>>>(x2, ffn_norm, hb, DMODEL);
  gemm_bf16_wmma<false><<<dim3(FFNH / BN, MTOK / BM), blk, 0, stream>>>(
      hb, w_fg, nullptr, gf, MTOK, FFNH, DMODEL);
  gemm_bf16_wmma<false><<<dim3(FFNH / BN, MTOK / BM), blk, 0, stream>>>(
      hb, w_fu, nullptr, uf, MTOK, FFNH, DMODEL);
  silu_mul<<<(MTOK * FFNH + 255) / 256, blk, 0, stream>>>(
      gf, uf, tb, MTOK * FFNH);
  gemm_bf16_wmma<true><<<dim3(DMODEL / BN, MTOK / BM), blk, 0, stream>>>(
      tb, w_fd, x2, out, MTOK, DMODEL, FFNH);
}
